// AttentionChromaSplit_90426241450270
// MI455X (gfx1250) — matrix operand order repaired and checked by reading, not compiled, not run on hardware
//
#include <hip/hip_runtime.h>
#include <hip/hip_bf16.h>
#include <stdint.h>

// Problem constants (from reference)
#define BB_TOT 16          // B*C
#define T_LEN  517
#define FEAT_D 120
#define NCH    3072        // N*K channels
#define NHEAD  24          // K attention heads
#define HDIM   128         // N per head
#define BT_TOT (BB_TOT * T_LEN)   // 8272 flattened (bb,t)
#define BT_PAD 8320        // 65*128: padded row count so stores need no guards

typedef __attribute__((ext_vector_type(16))) _Float16 v16h;
typedef __attribute__((ext_vector_type(8)))  float    v8f;

union F16Frag { v16h v; uint4 q[2]; _Float16 h[16]; };
union PackH   { uint4 u; _Float16 h[8]; };

static __device__ __forceinline__ v8f wmma_f16(v16h a, v16h b, v8f c) {
  // D = A(16x32 f16) * B(32x16 f16) + C(16x16 f32)
  return __builtin_amdgcn_wmma_f32_16x16x32_f16(false, a, false, b, (short)0, c,
                                                false, false);
}

// ---------------------------------------------------------------------------
// Kernel 1: QKV projection GEMM.  M = channel, N = bt (time), K = feature
// (120 padded to 128).  With cols = bt, a lane's 8 D-rows are 8 consecutive
// channels of ONE bt row -> one packed global_store_b128 per tile.
// nt-tiles processed in pairs with independent acc chains so WMMA hazard
// windows are filled by the sibling chain instead of v_nops.
// ---------------------------------------------------------------------------
__global__ __launch_bounds__(256) void k1_qkv(
    const float* __restrict__ x,
    const float* __restrict__ Wq, const float* __restrict__ bq,
    const float* __restrict__ Wk, const float* __restrict__ bk,
    const float* __restrict__ Wv, const float* __restrict__ bv,
    _Float16* __restrict__ qkv)
{
  extern __shared__ __align__(32) char smem[];
  _Float16* xRaw = (_Float16*)smem;             // [128 f][128 bt]
  _Float16* xB   = (_Float16*)(smem + 32768);   // B-frag layout [kk][nt][lane][16]

  const int tid  = threadIdx.x;
  const int lane = tid & 31, w = tid >> 5;
  const int g = lane >> 4, lm = lane & 15;
  const int bt0 = blockIdx.x * 128;

  // Stage x tile (coalesced over t), f padded to 128 with zeros.
  for (int it = 0; it < 64; ++it) {
    int lin = tid + it * 256;                // 0..16383
    int f = lin >> 7, col = lin & 127;
    int bt = bt0 + col;
    float v = 0.f;
    if (bt < BT_TOT && f < FEAT_D) {
      int bb = bt / T_LEN, tt = bt - bb * T_LEN;
      v = x[(size_t)(bb * FEAT_D + f) * T_LEN + tt];
    }
    xRaw[f * 128 + col] = (_Float16)v;
  }
  __syncthreads();
  // Permute into WMMA B-fragment layout: element j of lane L holds
  // B[k = 8*(L>>4) + (j&7) + 16*(j>>3)][n = L%16]  (CDNA5 16-bit B 32x16 layout).
  for (int it = 0; it < 64; ++it) {
    int lin = tid + it * 256;
    int j = lin & 15, L = (lin >> 4) & 31, nt = (lin >> 9) & 7, kk = lin >> 12;
    int f = kk * 32 + 8 * (L >> 4) + (j & 7) + 16 * (j >> 3);
    int btl = nt * 16 + (L & 15);
    xB[lin] = xRaw[f * 128 + btl];
  }
  __syncthreads();

  const float* Ws[3] = {Wq, Wk, Wv};
  const float* Bs[3] = {bq, bk, bv};

  // 576 channel-tile columns (3 mats * 192) split: 4 block-groups * 8 waves * 18
  for (int i = 0; i < 18; ++i) {
    int tc = blockIdx.y * 144 + w * 18 + i;
    int mat = tc / 192, ct = tc - mat * 192;
    const float* W = Ws[mat];
    int c = ct * 16 + lm;                    // A-row this lane owns

    // Prefetch next tile's weight row into cache while this tile computes.
    {
      int tcn = tc + ((i + 1 < 18) ? 1 : 0);
      int matn = tcn / 192, ctn = tcn - matn * 192;
      const float* Wn = Ws[matn] + (size_t)(ctn * 16 + lm) * FEAT_D;
      __builtin_prefetch(Wn, 0, 1);
      __builtin_prefetch(Wn + 64, 0, 1);
    }

    // A-fragment (weights): elem j holds f = kk*32 + 8*g + 16*(j>>3) + (j&7);
    // two contiguous 8-float groups.  Only (kk==3, g==1, j>=8) is padded.
    F16Frag Af[4];
    const float* rowp = W + (size_t)c * FEAT_D;
#pragma unroll
    for (int kk = 0; kk < 4; ++kk) {
      int f0 = kk * 32 + 8 * g;              // group0: j = 0..7
      int f1 = f0 + 16;                      // group1: j = 8..15
#pragma unroll
      for (int j = 0; j < 8; ++j)
        Af[kk].h[j] = (_Float16)rowp[f0 + j];
      if (kk == 3 && g == 1) {
#pragma unroll
        for (int j = 0; j < 8; ++j) Af[kk].h[8 + j] = (_Float16)0.f;
      } else {
#pragma unroll
        for (int j = 0; j < 8; ++j) Af[kk].h[8 + j] = (_Float16)rowp[f1 + j];
      }
    }
    // Bias for this lane's 8 D-rows: channels ct*16 + 8g + 0..7 (contiguous).
    const float* bp = Bs[mat] + ct * 16 + 8 * g;
    v8f binit = {bp[0], bp[1], bp[2], bp[3], bp[4], bp[5], bp[6], bp[7]};
    _Float16* outp = qkv + (size_t)mat * BT_PAD * NCH;

    for (int nt = 0; nt < 8; nt += 2) {
      v8f acc0 = binit, acc1 = binit;        // two independent WMMA chains
#pragma unroll
      for (int kk = 0; kk < 4; ++kk) {
        v16h B0 = *(const v16h*)(xB + ((kk * 8 + nt)     * 32 + lane) * 16);
        v16h B1 = *(const v16h*)(xB + ((kk * 8 + nt + 1) * 32 + lane) * 16);
        acc0 = wmma_f16(Af[kk].v, B0, acc0);
        acc1 = wmma_f16(Af[kk].v, B1, acc1);
      }
      // Pack 8 consecutive channels of one bt row -> single b128 store each.
      PackH o0, o1;
#pragma unroll
      for (int r = 0; r < 8; ++r) { o0.h[r] = (_Float16)acc0[r];
                                    o1.h[r] = (_Float16)acc1[r]; }
      size_t cb = (size_t)ct * 16 + 8 * g;
      int btA = bt0 + nt * 16 + lm;
      *(uint4*)(outp + (size_t)btA * NCH + cb)        = o0.u;
      *(uint4*)(outp + (size_t)(btA + 16) * NCH + cb) = o1.u;
    }
  }
}

// ---------------------------------------------------------------------------
// Kernel 2: per (bb, 16-time tile) attention, all-WMMA.
// logits[t,i,j] = sum_n q[t,i,n]k[t,j,n]  -> softmax over j (24)
// out[t,i,n]    = sum_j wei[t,i,j]v[t,j,n]   (A = v^T, B = wei^T)
// q/k staged via CDNA5 async global->LDS copies (ASYNCcnt).
// ---------------------------------------------------------------------------
#define K2_OFF_QS    0          // q [16][3072] f16 (96K); later weiB overlay
#define K2_OFF_KS    98304      // k [16][3072] f16; later vt[16][128][32] (128K)
#define K2_OFF_WEI   229376     // logits f32 [16][32][32] (64K)
#define K2_OFF_STATS 294912     // [24][2] f32
#define K2_SMEM_BYTES 295168

__global__ __launch_bounds__(256) void k2_attn(
    const _Float16* __restrict__ qkv,
    float* __restrict__ att, float* __restrict__ stats_out)
{
  extern __shared__ __align__(32) char smem[];
  _Float16* qs   = (_Float16*)(smem + K2_OFF_QS);
  _Float16* ks   = (_Float16*)(smem + K2_OFF_KS);
  _Float16* weiB = (_Float16*)(smem + K2_OFF_QS);     // [16][2][32][16] f16
  float*    wei  = (float*)(smem + K2_OFF_WEI);
  float*    st   = (float*)(smem + K2_OFF_STATS);

  const int tid = threadIdx.x, lane = tid & 31, w = tid >> 5;
  const int g = lane >> 4, lm = lane & 15;
  const int bb = blockIdx.x / 33, tt = blockIdx.x - bb * 33;
  const int t0 = tt * 16;

  if (tid < 48) st[tid] = 0.f;

  const _Float16* qg = qkv;
  const _Float16* kg = qkv + (size_t)BT_PAD * NCH;
  const _Float16* vg = qkv + (size_t)2 * BT_PAD * NCH;

  // Stage q,k rows with async global->LDS b128 copies; zero out-of-range rows.
  for (int it = 0; it < 24; ++it) {
    int u = tid + it * 256;                  // 0..6143 uint4s
    int row = u / 384, coff = (u - row * 384) * 8;
    int tglob = t0 + row;
    if (tglob < T_LEN) {
      size_t base = (size_t)(bb * T_LEN + tglob) * NCH + coff;
      uint32_t dq = (uint32_t)(uintptr_t)(qs + row * NCH + coff);
      uint32_t dk = (uint32_t)(uintptr_t)(ks + row * NCH + coff);
      uint64_t aq = (uint64_t)(uintptr_t)(qg + base);
      uint64_t ak = (uint64_t)(uintptr_t)(kg + base);
      asm volatile("global_load_async_to_lds_b128 %0, %1, off\n\t"
                   "global_load_async_to_lds_b128 %2, %3, off"
                   :: "v"(dq), "v"(aq), "v"(dk), "v"(ak) : "memory");
    } else {
      *(uint4*)(qs + row * NCH + coff) = make_uint4(0, 0, 0, 0);
      *(uint4*)(ks + row * NCH + coff) = make_uint4(0, 0, 0, 0);
    }
  }
  asm volatile("s_wait_asynccnt 0x0" ::: "memory");
  __syncthreads();

  // Phase 1: logits via WMMA, K = n (128 -> 4 chunks of 32).
  // Two j-column tiles run as interleaved independent acc chains.
  // jcol = lm is always a valid head; jcol = 16+lm valid iff lm<8.
  for (int ti = 0; ti < 2; ++ti) {
    int t = w * 2 + ti;
#pragma unroll
    for (int mt = 0; mt < 2; ++mt) {
      F16Frag A[4];
      bool aval = (mt == 0) || (lm < 8);
      int irow = mt * 16 + lm;
#pragma unroll
      for (int kk = 0; kk < 4; ++kk) {
        if (aval) {
          const _Float16* p = qs + t * NCH + irow * HDIM + kk * 32 + 16 * g;
          A[kk].q[0] = *(const uint4*)p;
          A[kk].q[1] = *(const uint4*)(p + 8);
        } else {
          A[kk].q[0] = make_uint4(0, 0, 0, 0);
          A[kk].q[1] = make_uint4(0, 0, 0, 0);
        }
      }
      v8f acc0 = {0, 0, 0, 0, 0, 0, 0, 0};
      v8f acc1 = acc0;
#pragma unroll
      for (int kk = 0; kk < 4; ++kk) {
        v16h B0 = *(const v16h*)(ks + t * NCH + lm * HDIM + kk * 32 + 16 * g);
        F16Frag B1;
        if (lm < 8) {
          B1.v = *(const v16h*)(ks + t * NCH + (16 + lm) * HDIM + kk * 32 + 16 * g);
        } else {
          B1.q[0] = make_uint4(0, 0, 0, 0); B1.q[1] = B1.q[0];
        }
        acc0 = wmma_f16(A[kk].v, B0, acc0);
        acc1 = wmma_f16(A[kk].v, B1.v, acc1);
      }
#pragma unroll
      for (int r = 0; r < 8; ++r) {
        int i = mt * 16 + r + 8 * g;
        wei[(t * 32 + i) * 32 + lm]      = acc0[r];
        wei[(t * 32 + i) * 32 + 16 + lm] = acc1[r];
      }
    }
  }
  __syncthreads();

  // Phase 2a: stage v transposed for A-fragments: vt[t][n][head(32)]
  _Float16* vt = ks;
  for (int it = 0; it < 24; ++it) {
    int u = tid + it * 256;
    int row = u / 384, coff = (u - row * 384) * 8;
    int tglob = t0 + row;
    F16Frag vv; vv.q[0] = make_uint4(0, 0, 0, 0);
    if (tglob < T_LEN)
      vv.q[0] = *(const uint4*)(vg + (size_t)(bb * T_LEN + tglob) * NCH + coff);
    int head = coff >> 7, n = coff & 127;
#pragma unroll
    for (int e = 0; e < 8; ++e)
      vt[(row * 128 + n + e) * 32 + head] = vv.h[e];
  }
  // zero head-pad 24..31 (so 0-weights never multiply NaN garbage)
  for (int it = 0; it < 8; ++it) {
    int u = tid + it * 256;                  // (t,n)
    int row = u >> 7, n = u & 127;
    *(uint4*)(vt + (row * 128 + n) * 32 + 24) = make_uint4(0, 0, 0, 0);
  }

  // Phase 2b: softmax rows -> weiB = wei^T in B-fragment layout:
  // lane L of tile (t,mt): col i = mt*16 + L%16, elem j: head k = j + 16*(L>>4)
  for (int rr = tid; rr < 512; rr += 256) {
    int t = rr >> 5, i = rr & 31;
    float p[32];
    if (i < NHEAD) {
      float mx = -1e30f;
#pragma unroll
      for (int j = 0; j < NHEAD; ++j) {
        p[j] = wei[(t * 32 + i) * 32 + j];
        mx = fmaxf(mx, p[j]);
      }
      float s = 0.f;
#pragma unroll
      for (int j = 0; j < NHEAD; ++j) { p[j] = __expf(p[j] - mx); s += p[j]; }
      float inv = 1.f / s;
#pragma unroll
      for (int j = 0; j < NHEAD; ++j) p[j] *= inv;
#pragma unroll
      for (int j = NHEAD; j < 32; ++j) p[j] = 0.f;
    } else {
#pragma unroll
      for (int j = 0; j < 32; ++j) p[j] = 0.f;
    }
    int mt = i >> 4, m = i & 15;
#pragma unroll
    for (int gg = 0; gg < 2; ++gg)
#pragma unroll
      for (int j = 0; j < 16; ++j)
        weiB[((t * 2 + mt) * 32 + gg * 16 + m) * 16 + j] = (_Float16)p[j + 16 * gg];
  }
  __syncthreads();

  // Phase 3: out = (v^T) x (wei^T): A rows = n, B cols = head, K = 32 heads.
  // A lane's 8 D-rows are 8 consecutive n of ONE head -> 2x b128 f32 stores.
  // The two head-tiles are independent acc chains per nt.
  float ss0 = 0.f, sq0 = 0.f, ss1 = 0.f, sq1 = 0.f;

  for (int ti = 0; ti < 2; ++ti) {
    int t = w * 2 + ti;
    bool tvalid = (t0 + t) < T_LEN;          // wave-uniform
    size_t btg = (size_t)(bb * T_LEN + t0 + t);
    v16h B0 = *(const v16h*)(weiB + ((t * 2 + 0) * 32 + lane) * 16);
    v16h B1 = *(const v16h*)(weiB + ((t * 2 + 1) * 32 + lane) * 16);
    for (int nt = 0; nt < 8; ++nt) {
      F16Frag A;
      const _Float16* ap = vt + (size_t)(t * 128 + nt * 16 + lm) * 32 + 16 * g;
      A.q[0] = *(const uint4*)ap;
      A.q[1] = *(const uint4*)(ap + 8);
      // n values this lane covers: nt*16 + 8g + 0..7 (contiguous)
      size_t nbase = (size_t)nt * 16 + 8 * g;

      v8f a0 = {0, 0, 0, 0, 0, 0, 0, 0};
      v8f a1 = a0;
      a0 = wmma_f16(A.v, B0, a0);
      a1 = wmma_f16(A.v, B1, a1);
      {
        float* op = att + btg * NCH + (size_t)lm * HDIM + nbase;
        float4 lo = {a0[0], a0[1], a0[2], a0[3]};
        float4 hi = {a0[4], a0[5], a0[6], a0[7]};
        *(float4*)op = lo;  *(float4*)(op + 4) = hi;
      }
      if (lm < 8) {
        float* op = att + btg * NCH + (size_t)(16 + lm) * HDIM + nbase;
        float4 lo = {a1[0], a1[1], a1[2], a1[3]};
        float4 hi = {a1[4], a1[5], a1[6], a1[7]};
        *(float4*)op = lo;  *(float4*)(op + 4) = hi;
      }
      if (tvalid) {
#pragma unroll
        for (int r = 0; r < 8; ++r) {
          ss0 += a0[r]; sq0 += a0[r] * a0[r];
          ss1 += a1[r]; sq1 += a1[r] * a1[r];
        }
      }
    }
  }
  atomicAdd(&st[lm * 2],     ss0);
  atomicAdd(&st[lm * 2 + 1], sq0);
  if (lm < 8) {                 // heads 16..23 (lanes lm>=8 computed zeros)
    atomicAdd(&st[(16 + lm) * 2],     ss1);
    atomicAdd(&st[(16 + lm) * 2 + 1], sq1);
  }
  __syncthreads();
  if (tid < NHEAD) {
    size_t o = ((size_t)(bb * 33 + tt) * NHEAD + tid) * 2;
    stats_out[o]     = st[tid * 2];
    stats_out[o + 1] = st[tid * 2 + 1];
  }
}

// ---------------------------------------------------------------------------
// Kernel 3: reduce LN partials, normalize, scale, and transpose-write the
// final [BB][N][K][T] output.  Block = (bb, head, 32-n tile, 128-t tile).
// ---------------------------------------------------------------------------
__global__ __launch_bounds__(256) void k3_ln(
    const float* __restrict__ att, const float* __restrict__ stats,
    const float* __restrict__ gamma, const float* __restrict__ beta,
    float* __restrict__ out)
{
  extern __shared__ __align__(16) char smem[];
  float* ts  = (float*)smem;                              // [32][129]
  float* gs  = (float*)(smem + 32 * 129 * 4);             // [32][128]
  float* bsh = (float*)(smem + 32 * 129 * 4 + 32 * 128 * 4);
  float* red = (float*)(smem + 32 * 129 * 4 + 2 * 32 * 128 * 4);

  int b = blockIdx.x;
  int tti = b % 5;  b /= 5;
  int nti = b % 4;  b /= 4;
  int k   = b % 24; int bb = b / 24;
  int t0 = tti * 128, n0 = nti * 32;
  int tid = threadIdx.x;

  if (tid < 2) red[tid] = 0.f;
  __syncthreads();
  if (tid < 33) {
    const float* sp = stats + ((size_t)(bb * 33 + tid) * NHEAD + k) * 2;
    atomicAdd(&red[0], sp[0]);
    atomicAdd(&red[1], sp[1]);
  }
  __syncthreads();
  const float cnt = (float)(T_LEN * HDIM);
  float mean = red[0] / cnt;
  float var  = red[1] / cnt - mean * mean;
  float rstd = rsqrtf(var + 1e-5f);

  for (int it = 0; it < 16; ++it) {
    int u = tid + it * 256;                 // gamma/beta: coalesced over t
    int nn = u >> 7, tl = u & 127;
    int t = t0 + tl;
    float gv = 0.f, bv = 0.f;
    if (t < T_LEN) {
      gv = gamma[(size_t)(n0 + nn) * T_LEN + t];
      bv = beta[(size_t)(n0 + nn) * T_LEN + t];
    }
    gs[nn * 128 + tl] = gv;
    bsh[nn * 128 + tl] = bv;
  }
  for (int it = 0; it < 16; ++it) {
    int u = tid + it * 256;                 // att: coalesced over n
    int tl = u >> 5, nn = u & 31;
    int t = t0 + tl;
    float v = 0.f;
    if (t < T_LEN)
      v = att[(size_t)(bb * T_LEN + t) * NCH + k * HDIM + n0 + nn];
    ts[nn * 129 + tl] = v;
  }
  __syncthreads();
  for (int it = 0; it < 16; ++it) {
    int u = tid + it * 256;                 // out: coalesced over t
    int nn = u >> 7, tl = u & 127;
    int t = t0 + tl;
    if (t < T_LEN) {
      float v = (ts[nn * 129 + tl] - mean) * rstd * gs[nn * 128 + tl]
                + bsh[nn * 128 + tl];
      out[((size_t)(bb * HDIM + n0 + nn) * NHEAD + k) * T_LEN + t] = v;
    }
  }
}

// ---------------------------------------------------------------------------
extern "C" void kernel_launch(void* const* d_in, const int* in_sizes, int n_in,
                              void* d_out, int out_size, void* d_ws, size_t ws_size,
                              hipStream_t stream) {
  const float* x     = (const float*)d_in[0];
  const float* Wq    = (const float*)d_in[1];
  const float* bq    = (const float*)d_in[2];
  const float* Wk    = (const float*)d_in[3];
  const float* bk    = (const float*)d_in[4];
  const float* Wv    = (const float*)d_in[5];
  const float* bv    = (const float*)d_in[6];
  const float* gamma = (const float*)d_in[7];
  const float* beta  = (const float*)d_in[8];
  float* out = (float*)d_out;

  // Workspace: QKV f16 (padded rows) | pre-LN att f32 (padded rows) | partials
  _Float16* qkv = (_Float16*)d_ws;
  size_t qkvBytes = (size_t)3 * BT_PAD * NCH * sizeof(_Float16);
  float* att = (float*)((char*)d_ws + qkvBytes);
  size_t attBytes = (size_t)BT_PAD * NCH * sizeof(float);
  float* stats = (float*)((char*)d_ws + qkvBytes + attBytes);

  k1_qkv<<<dim3(65, 4, 1), 256, 65536, stream>>>(x, Wq, bq, Wk, bk, Wv, bv, qkv);
  k2_attn<<<dim3(16 * 33, 1, 1), 256, K2_SMEM_BYTES, stream>>>(qkv, att, stats);
  k3_ln<<<dim3(16 * 24 * 4 * 5, 1, 1), 256,
          32 * 129 * 4 + 2 * 32 * 128 * 4 + 16, stream>>>(att, stats, gamma,
                                                          beta, out);
}
